// Attention_46926812676211
// MI455X (gfx1250) — compile-verified, matrix operand order checked by reading, not run on hardware
//
#include <hip/hip_runtime.h>
#include <hip/hip_bf16.h>

typedef __attribute__((ext_vector_type(16))) _Float16 v16h;
typedef __attribute__((ext_vector_type(8)))  _Float16 v8h;
typedef __attribute__((ext_vector_type(8)))  float    v8f;
typedef __attribute__((ext_vector_type(4)))  unsigned u32x4;
typedef __attribute__((ext_vector_type(8)))  unsigned u32x8;

// Problem constants (match reference)
constexpr int B_  = 4;
constexpr int L_  = 2048;
constexpr int DM  = 1024;      // d_model
constexpr int H_  = 16;        // heads
constexpr int DK  = 64;        // head dim
constexpr int BL  = B_ * L_;   // 8192 rows
constexpr float SCALE = 0.125f; // 1/sqrt(64)
constexpr float NEG_BIG = -1e30f;

// ---------------------------------------------------------------------------
// WMMA helper (gfx1250, wave32). D = A(16x32 f16) * B(32x16 f16) + C(16x16 f32)
// ---------------------------------------------------------------------------
__device__ __forceinline__ v8f wmma_f16(v16h a, v16h b, v8f c) {
  return __builtin_amdgcn_wmma_f32_16x16x32_f16(
      /*neg_a=*/false, a, /*neg_b=*/false, b,
      /*c_mod=*/(short)0, c, /*reuse_a=*/false, /*reuse_b=*/false);
}

// Load a 16x32 f16 fragment whose 16 "rows" (A: M, B: N via B^T) live at
// rows r0..r0+15 of a row-major matrix with leading dim `ld` (halves),
// K-window [k0, k0+32).  ISA 16-bit A layout: lane = row%16,
// halves j=0..7  -> K = k0 + 8*(lane/16) + j
// halves j=8..15 -> K = k0 + 16 + 8*(lane/16) + (j-8)
// Both 8-half groups are contiguous -> two 16-byte loads per lane.
__device__ __forceinline__ v16h load_frag16(const _Float16* p, int ld, int r0, int k0) {
  const int lane = threadIdx.x & 31;
  const _Float16* rp = p + (size_t)(r0 + (lane & 15)) * ld + k0 + ((lane >> 4) << 3);
  v8h lo = *(const v8h*)rp;
  v8h hi = *(const v8h*)(rp + 16);
  v16h f;
#pragma unroll
  for (int j = 0; j < 8; ++j) { f[j] = lo[j]; f[j + 8] = hi[j]; }
  return f;
}

// Row-wise reduction across the 16-lane half-group (C/D layout: N = lane%16).
__device__ __forceinline__ float rowmax16(float v) {
#pragma unroll
  for (int m = 1; m < 16; m <<= 1) v = fmaxf(v, __shfl_xor(v, m, 32));
  return v;
}
__device__ __forceinline__ float rowsum16(float v) {
#pragma unroll
  for (int m = 1; m < 16; m <<= 1) v += __shfl_xor(v, m, 32);
  return v;
}

// ---------------------------------------------------------------------------
// TDM: issue tensor_load_to_lds for a 2D f16 tile.
// Tensor: row-major [tensor_d1 x tensor_d0], row stride `stride0` (elements).
// Tile:   [tile_d1 rows x tile_d0 elements] starting at `gptr`, packed
//         row-major into LDS at byte offset `lds_addr`.
// D# packing per CDNA5 ISA (2D: groups 2/3 = NULL).
// All inputs are workgroup-uniform -> SGPRs; issued by one wave (EXEC ignored),
// completion tracked with TENSORcnt (in-order per wave).
// ---------------------------------------------------------------------------
__device__ __forceinline__ void tdm_load_2d_f16(unsigned lds_addr, const void* gptr,
                                                unsigned tensor_d0, unsigned tensor_d1,
                                                unsigned stride0,
                                                unsigned tile_d0, unsigned tile_d1) {
  const unsigned long long ga = (unsigned long long)(uintptr_t)gptr;
  u32x4 g0;
  g0[0] = 1u;                                           // count=1, user mode
  g0[1] = lds_addr;                                     // LDS byte address
  g0[2] = (unsigned)(ga & 0xFFFFFFFFu);                 // global_addr[31:0]
  g0[3] = (unsigned)((ga >> 32) & 0x01FFFFFFu)          // global_addr[56:32]
          | (2u << 30);                                 // type=2 ("image")
  u32x8 g1;
  g1[0] = 1u << 16;                                     // data_size=1 (2B), wg_mask=0
  g1[1] = (tensor_d0 & 0xFFFFu) << 16;                  // tensor_dim0[15:0]
  g1[2] = ((tensor_d0 >> 16) & 0xFFFFu)                 // tensor_dim0[31:16]
          | ((tensor_d1 & 0xFFFFu) << 16);              // tensor_dim1[15:0]
  g1[3] = ((tensor_d1 >> 16) & 0xFFFFu)                 // tensor_dim1[31:16]
          | ((tile_d0 & 0xFFFFu) << 16);                // tile_dim0
  g1[4] = tile_d1 & 0xFFFFu;                            // tile_dim1 (tile_dim2=0)
  g1[5] = stride0;                                      // tensor_dim0_stride[31:0]
  g1[6] = 0u;                                           // stride0[47:32] | stride1 lo
  g1[7] = 0u;                                           // stride1 hi
  asm volatile("tensor_load_to_lds %0, %1" :: "s"(g0), "s"(g1) : "memory");
}

// ---------------------------------------------------------------------------
// 1) fp32 -> f16 conversion
// ---------------------------------------------------------------------------
__global__ void k_f32_to_f16(const float* __restrict__ src, _Float16* __restrict__ dst, int n) {
  int i = blockIdx.x * blockDim.x + threadIdx.x;
  if (i < n) dst[i] = (_Float16)src[i];
}

// ---------------------------------------------------------------------------
// 2) LayerNorm (one 256-thread block per row of 1024), fp32 math, f16 output
// ---------------------------------------------------------------------------
__global__ void k_layernorm(const float* __restrict__ x, _Float16* __restrict__ h) {
  __shared__ float s1[256], s2[256];
  const int row = blockIdx.x, tid = threadIdx.x;
  const float* xr = x + (size_t)row * DM;
  float vals[4], s = 0.f, sq = 0.f;
#pragma unroll
  for (int j = 0; j < 4; ++j) {
    float v = xr[tid + j * 256];
    vals[j] = v; s += v; sq += v * v;
  }
  s1[tid] = s; s2[tid] = sq;
  __syncthreads();
  for (int off = 128; off > 0; off >>= 1) {
    if (tid < off) { s1[tid] += s1[tid + off]; s2[tid] += s2[tid + off]; }
    __syncthreads();
  }
  const float mean = s1[0] * (1.f / DM);
  const float var  = s2[0] * (1.f / DM) - mean * mean;
  const float rstd = rsqrtf(var + 1e-8f);
  _Float16* hr = h + (size_t)row * DM;
#pragma unroll
  for (int j = 0; j < 4; ++j)
    hr[tid + j * 256] = (_Float16)((vals[j] - mean) * rstd);
}

// ---------------------------------------------------------------------------
// 3/6) WMMA GEMM with TDM-fed LDS: C[M,N] = A[M,K] @ W[N,K]^T.
//   Block = 128 threads (4 waves), 128x64 output tile; wave owns 32x64.
//   W tile (64 n-rows x 32 k) is DMA'd into LDS by the Tensor Data Mover,
//   double-buffered one k-step ahead (wave 0 issues, s_wait_tensorcnt gates,
//   barriers publish).  A fragments stream direct from global (per-wave rows).
//   M must be a multiple of 128, N of 64, K of 32.
// ---------------------------------------------------------------------------
__global__ void k_gemm_aWt_tdm(const _Float16* __restrict__ A, const _Float16* __restrict__ W,
                               float* __restrict__ Cf32, _Float16* __restrict__ Cf16,
                               int M, int N, int K) {
  __shared__ __align__(16) _Float16 sW[2][64 * 32];   // 8 KB double buffer
  const int lane = threadIdx.x & 31;
  const int wave = threadIdx.x >> 5;
  const int n0 = blockIdx.x * 64;
  const int m0 = blockIdx.y * 128 + wave * 32;
  const bool tdm_wave = (threadIdx.x < 32);           // wave-uniform

  v8f acc[2][4];
#pragma unroll
  for (int hh = 0; hh < 2; ++hh)
#pragma unroll
    for (int t = 0; t < 4; ++t) acc[hh][t] = (v8f){0, 0, 0, 0, 0, 0, 0, 0};

  const int nsteps = K / 32;
  // Prologue: DMA the first W tile.
  if (tdm_wave)
    tdm_load_2d_f16((unsigned)(uintptr_t)&sW[0][0],
                    W + (size_t)n0 * K,
                    (unsigned)K, (unsigned)N, (unsigned)K,
                    /*tile_d0=*/32u, /*tile_d1=*/64u);

  for (int s = 0; s < nsteps; ++s) {
    const int k0 = s * 32;
    __syncthreads();  // buffer (s+1)&1 is no longer being read (step s-1 done)
    if (tdm_wave) {
      if (s + 1 < nsteps) {
        tdm_load_2d_f16((unsigned)(uintptr_t)&sW[(s + 1) & 1][0],
                        W + (size_t)n0 * K + (size_t)(s + 1) * 32,
                        (unsigned)K, (unsigned)N, (unsigned)K, 32u, 64u);
        __builtin_amdgcn_s_wait_tensorcnt(1);  // current tile landed; next in flight
      } else {
        __builtin_amdgcn_s_wait_tensorcnt(0);
      }
    }
    __syncthreads();  // publish current tile to all waves

    // prefetch next A k-chunk through L2
    if (s + 1 < nsteps)
      __builtin_prefetch(A + (size_t)(m0 + lane) * K + k0 + 32, 0, 1);

    const v16h a0 = load_frag16(A, K, m0, k0);
    const v16h a1 = load_frag16(A, K, m0 + 16, k0);
    const _Float16* wbuf = &sW[s & 1][0];
#pragma unroll
    for (int t = 0; t < 4; ++t) {
      const v16h bf = load_frag16(wbuf, 32, t * 16, 0);   // ds_load_b128 x2
      acc[0][t] = wmma_f16(a0, bf, acc[0][t]);
      acc[1][t] = wmma_f16(a1, bf, acc[1][t]);
    }
  }

#pragma unroll
  for (int hh = 0; hh < 2; ++hh) {
    const int mrow = m0 + hh * 16 + 8 * (lane >> 4);
#pragma unroll
    for (int t = 0; t < 4; ++t) {
#pragma unroll
      for (int r = 0; r < 8; ++r) {
        const size_t idx = (size_t)(mrow + r) * N + (n0 + t * 16 + (lane & 15));
        if (Cf16) Cf16[idx] = (_Float16)acc[hh][t][r];
        else      Cf32[idx] = acc[hh][t][r];
      }
    }
  }
}

// ---------------------------------------------------------------------------
// 4) RoPE + repack.  qkv: [BL, 3*DM] f16 (q|k|v sections).
//    Q,K -> head-major [B*H, L, DK] f16 (rotated, fp32 trig)
//    V   -> transposed [B*H, DK, L] f16 (so P@V B-fragments are contiguous)
// ---------------------------------------------------------------------------
__global__ void k_rope_repack(const _Float16* __restrict__ qkv,
                              _Float16* __restrict__ Qh, _Float16* __restrict__ Kh,
                              _Float16* __restrict__ Vt) {
  const int idx = blockIdx.x * blockDim.x + threadIdx.x;
  if (idx >= B_ * H_ * L_ * (DK / 2)) return;
  const int i   = idx & 31;
  const int pos = (idx >> 5) & (L_ - 1);
  const int h   = (idx >> 16) & (H_ - 1);
  const int b   = idx >> 20;

  const _Float16* row = qkv + (size_t)(b * L_ + pos) * (3 * DM);
  const int c = h * DK + 2 * i;
  const float q1 = (float)row[c],          q2 = (float)row[c + 1];
  const float k1 = (float)row[DM + c],     k2 = (float)row[DM + c + 1];
  const float v1 = (float)row[2 * DM + c], v2 = (float)row[2 * DM + c + 1];

  const float inv = __expf(-(2.f * (float)i / (float)DK) * __logf(10000.f));
  const float ang = (float)pos * inv;
  const float cs = __cosf(ang), sn = __sinf(ang);

  const size_t base = ((size_t)(b * H_ + h) * L_ + pos) * DK;
  Qh[base + 2 * i]     = (_Float16)(q1 * cs - q2 * sn);
  Qh[base + 2 * i + 1] = (_Float16)(q1 * sn + q2 * cs);
  Kh[base + 2 * i]     = (_Float16)(k1 * cs - k2 * sn);
  Kh[base + 2 * i + 1] = (_Float16)(k1 * sn + k2 * cs);

  const size_t vb = (size_t)(b * H_ + h) * DK * L_;
  Vt[vb + (size_t)(2 * i) * L_ + pos]     = v1;
  Vt[vb + (size_t)(2 * i + 1) * L_ + pos] = v2;
}

// ---------------------------------------------------------------------------
// 5) Flash attention.  grid = (L/16 q-tiles, B*H), block = 1 wave (32 lanes).
//    Per 32-key step: 4 WMMAs for S(16x32), fp32 online softmax,
//    P staged via LDS (C-layout -> A-fragment), 4 WMMAs for O += P@V.
// ---------------------------------------------------------------------------
__global__ void k_flash_attn(const _Float16* __restrict__ Q, const _Float16* __restrict__ K,
                             const _Float16* __restrict__ Vt, _Float16* __restrict__ O) {
  __shared__ __align__(16) _Float16 sP[16 * 32];

  const int lane = threadIdx.x & 31;
  const int q0 = blockIdx.x * 16;
  const int bh = blockIdx.y;
  const int b = bh / H_, h = bh % H_;
  const int mrow = 8 * (lane >> 4);
  const int ncol = lane & 15;

  const _Float16* Qp = Q  + (size_t)bh * L_ * DK;
  const _Float16* Kp = K  + (size_t)bh * L_ * DK;
  const _Float16* Vp = Vt + (size_t)bh * DK * L_;

  const v16h qf0 = load_frag16(Qp, DK, q0, 0);
  const v16h qf1 = load_frag16(Qp, DK, q0, 32);

  v8f oacc[4];
#pragma unroll
  for (int t = 0; t < 4; ++t) oacc[t] = (v8f){0, 0, 0, 0, 0, 0, 0, 0};
  float m_run[8], l_run[8];
#pragma unroll
  for (int r = 0; r < 8; ++r) { m_run[r] = NEG_BIG; l_run[r] = 0.f; }

  for (int k0 = 0; k0 < q0 + 16; k0 += 32) {
    v8f s0 = (v8f){0, 0, 0, 0, 0, 0, 0, 0};
    v8f s1 = (v8f){0, 0, 0, 0, 0, 0, 0, 0};
    {
      v16h kf;
      kf = load_frag16(Kp, DK, k0, 0);       s0 = wmma_f16(qf0, kf, s0);
      kf = load_frag16(Kp, DK, k0, 32);      s0 = wmma_f16(qf1, kf, s0);
      kf = load_frag16(Kp, DK, k0 + 16, 0);  s1 = wmma_f16(qf0, kf, s1);
      kf = load_frag16(Kp, DK, k0 + 16, 32); s1 = wmma_f16(qf1, kf, s1);
    }

    float p0[8], p1[8], corr[8];
#pragma unroll
    for (int r = 0; r < 8; ++r) {
      const int q = q0 + mrow + r;
      const int keyA = k0 + ncol;
      const float a  = (keyA      <= q) ? s0[r] * SCALE : NEG_BIG;
      const float bb = (keyA + 16 <= q) ? s1[r] * SCALE : NEG_BIG;
      float mx = rowmax16(fmaxf(a, bb));
      const float mnew = fmaxf(m_run[r], mx);
      const float pa = __expf(a - mnew);
      const float pb = __expf(bb - mnew);
      p0[r] = pa; p1[r] = pb;
      const float rs = rowsum16(pa + pb);
      corr[r] = __expf(m_run[r] - mnew);
      l_run[r] = l_run[r] * corr[r] + rs;
      m_run[r] = mnew;
    }
#pragma unroll
    for (int t = 0; t < 4; ++t)
#pragma unroll
      for (int r = 0; r < 8; ++r) oacc[t][r] *= corr[r];

#pragma unroll
    for (int r = 0; r < 8; ++r) {
      sP[(mrow + r) * 32 + ncol]      = (_Float16)p0[r];
      sP[(mrow + r) * 32 + 16 + ncol] = (_Float16)p1[r];
    }
    __syncthreads();

    const v16h pf = load_frag16(sP, 32, 0, 0);
#pragma unroll
    for (int t = 0; t < 4; ++t) {
      v16h vf = load_frag16(Vp, L_, t * 16, k0);
      oacc[t] = wmma_f16(pf, vf, oacc[t]);
    }
    __syncthreads();
  }

#pragma unroll
  for (int t = 0; t < 4; ++t) {
#pragma unroll
    for (int r = 0; r < 8; ++r) {
      const int q = q0 + mrow + r;
      const int d = t * 16 + ncol;
      O[(size_t)(b * L_ + q) * DM + h * DK + d] = (_Float16)(oacc[t][r] / l_run[r]);
    }
  }
}

// ---------------------------------------------------------------------------
// Launch
// ---------------------------------------------------------------------------
extern "C" void kernel_launch(void* const* d_in, const int* in_sizes, int n_in,
                              void* d_out, int out_size, void* d_ws, size_t ws_size,
                              hipStream_t stream) {
  const float* x    = (const float*)d_in[0];   // [B, L, DM]
  const float* W_in = (const float*)d_in[1];   // [3*DM, DM]
  const float* W_o  = (const float*)d_in[2];   // [DM, DM]
  float* out = (float*)d_out;                  // [B, L, DM]

  char* ws = (char*)d_ws;
  size_t off = 0;
  auto alloc16 = [&](size_t elems) { _Float16* p = (_Float16*)(ws + off); off += elems * sizeof(_Float16); return p; };

  _Float16* h16   = alloc16((size_t)BL * DM);        // LayerNorm output
  _Float16* Win16 = alloc16((size_t)3 * DM * DM);    // W_in f16
  _Float16* Wo16  = alloc16((size_t)DM * DM);        // W_o f16
  _Float16* qkv16 = alloc16((size_t)BL * 3 * DM);    // QKV projection
  _Float16* Qh    = alloc16((size_t)BL * DM);        // [B*H, L, DK]
  _Float16* Kh    = alloc16((size_t)BL * DM);        // [B*H, L, DK]
  _Float16* Vt    = alloc16((size_t)BL * DM);        // [B*H, DK, L]
  _Float16* O16   = alloc16((size_t)BL * DM);        // attention output, [BL, DM]

  {
    int n1 = 3 * DM * DM, n2 = DM * DM;
    k_f32_to_f16<<<(n1 + 255) / 256, 256, 0, stream>>>(W_in, Win16, n1);
    k_f32_to_f16<<<(n2 + 255) / 256, 256, 0, stream>>>(W_o, Wo16, n2);
  }
  k_layernorm<<<BL, 256, 0, stream>>>(x, h16);
  // QKV projection: [8192,1024] @ [3072,1024]^T -> f16
  k_gemm_aWt_tdm<<<dim3((3 * DM) / 64, BL / 128), 128, 0, stream>>>(
      h16, Win16, nullptr, qkv16, BL, 3 * DM, DM);
  {
    int n = B_ * H_ * L_ * (DK / 2);
    k_rope_repack<<<(n + 255) / 256, 256, 0, stream>>>(qkv16, Qh, Kh, Vt);
  }
  k_flash_attn<<<dim3(L_ / 16, B_ * H_), 32, 0, stream>>>(Qh, Kh, Vt, O16);
  // Output projection: [8192,1024] @ [1024,1024]^T -> fp32
  k_gemm_aWt_tdm<<<dim3(DM / 64, BL / 128), 128, 0, stream>>>(
      O16, Wo16, out, nullptr, BL, DM, DM);
}
